// CAPA_62886911148616
// MI455X (gfx1250) — matrix operand order repaired and checked by reading, not compiled, not run on hardware
//
#include <hip/hip_runtime.h>

// CAPA local-window attention for MI455X (gfx1250, wave32, WMMA + async-LDS).
// Pipeline:
//   kernel 1: q = 0.125*Wq*x_key, k = Wk*x_ref, v = Wv*x_ref  (f16, channel-last)
//   kernel 2: per-window flash attention, v_wmma_f32_16x16x32_f16,
//             double-buffered key/value staging via global_load_async_to_lds_b128
// Workspace: 3 * 2*256*256*64 f16 = 48 MB in d_ws.

typedef __attribute__((ext_vector_type(16))) _Float16 v16h;
typedef __attribute__((ext_vector_type(8)))  float    v8f;

#define BATCH   2
#define CH      64
#define HW      65536          // 256*256
#define SCW     11
#define NSIDE   24             // windows per side (264/11)
#define NWIN    (NSIDE*NSIDE)  // 576
#define NKEY    1089           // 33*33
#define NKPAD   1120           // padded to 35*32
#define NCHUNK  35

#define QLD 72                 // padded LDS row strides (f16 elems)
#define KLD 72
#define VLD 36

// ---- gfx1250 async global->LDS copy (ASYNCcnt) --------------------------
__device__ __forceinline__ void async_copy_b128(unsigned lds_off,
                                                unsigned long long gaddr) {
    asm volatile("global_load_async_to_lds_b128 %0, %1, off"
                 :: "v"(lds_off), "v"(gaddr) : "memory");
}
__device__ __forceinline__ void wait_asynccnt0() {
    asm volatile("s_wait_asynccnt 0" ::: "memory");
}

// ---------------------------------------------------------------------------
// Kernel 1: projections (WMMA GEMM, 64x64 weights x 64xP pixels)
// x layout: [b][c][65536] f32.  Output: channel-last f16 [b*65536][64].
// ---------------------------------------------------------------------------
__global__ __launch_bounds__(256) void capa_proj(
    const float* __restrict__ xk, const float* __restrict__ xr,
    const float* __restrict__ Wq, const float* __restrict__ Wk,
    const float* __restrict__ Wv,
    _Float16* __restrict__ qh, _Float16* __restrict__ kh,
    _Float16* __restrict__ vh)
{
    alignas(16) __shared__ _Float16 wA[3][64][QLD];
    const int t = threadIdx.x;

    for (int i = t; i < 64 * 64; i += 256) {
        int o = i >> 6, c = i & 63;
        wA[0][o][c] = (_Float16)(Wq[i] * 0.125f);   // fold softmax scale into q
        wA[1][o][c] = (_Float16)(Wk[i]);
        wA[2][o][c] = (_Float16)(Wv[i]);
    }
    __syncthreads();

    const int wave = t >> 5, lane = t & 31;
    const int low  = (lane < 16) ? 1 : 0;
    const int pix  = blockIdx.x * 128 + wave * 16 + (lane & 15);  // < 131072
    const int bb   = pix >> 16;
    const int idx  = pix & 65535;
    const long xbase = (long)bb * CH * HW + idx;

    // B fragments (K=channel, N=pixel) for x_key and x_ref, 2 k-steps of 32.
    v16h Bk[2], Br[2];
#pragma unroll
    for (int ks = 0; ks < 2; ++ks) {
        v16h fk, fr;
        int cb = ks * 32 + (low ? 0 : 16);
#pragma unroll
        for (int j = 0; j < 8; ++j) {
            long a0 = xbase + (long)(cb + 2 * j)     * HW;
            long a1 = xbase + (long)(cb + 2 * j + 1) * HW;
            fk[2 * j]     = (_Float16)xk[a0];
            fk[2 * j + 1] = (_Float16)xk[a1];
            fr[2 * j]     = (_Float16)xr[a0];
            fr[2 * j + 1] = (_Float16)xr[a1];
        }
        Bk[ks] = fk; Br[ks] = fr;
    }

#pragma unroll
    for (int mt = 0; mt < 4; ++mt) {
        const int m = mt * 16 + (lane & 15);
#pragma unroll
        for (int mat = 0; mat < 3; ++mat) {
            v8f acc = {};
#pragma unroll
            for (int ks = 0; ks < 2; ++ks) {
                v16h a;
                const _Float16* wrow = &wA[mat][m][0];
#pragma unroll
                for (int p = 0; p < 8; ++p) {
                    int K0 = ks * 32 + 2 * p
                             + (low ? 0 : 16);
                    a[2 * p]     = wrow[K0];
                    a[2 * p + 1] = wrow[K0 + 1];
                }
                v16h b = (mat == 0) ? Bk[ks] : Br[ks];
                acc = __builtin_amdgcn_wmma_f32_16x16x32_f16(
                        false, a, false, b, (short)0, acc, false, false);
            }
            _Float16* dst = (mat == 0) ? qh : (mat == 1) ? kh : vh;
            const int ch0 = mt * 16 + (low ? 0 : 8);
            const long ob = (long)pix * CH + ch0;
#pragma unroll
            for (int i = 0; i < 4; ++i) {
                _Float16 lo = (_Float16)acc[2 * i];
                _Float16 hi = (_Float16)acc[2 * i + 1];
                unsigned int pk =
                    ((unsigned int)__builtin_bit_cast(unsigned short, hi) << 16) |
                     (unsigned int)__builtin_bit_cast(unsigned short, lo);
                *(unsigned int*)(dst + ob + 2 * i) = pk;
            }
        }
    }
}

// ---------------------------------------------------------------------------
// Kernel 2: per-window flash attention, double-buffered async staging.
// One block = one (batch, window). 8 waves; wave w owns query tile w (16 q).
// ---------------------------------------------------------------------------
__global__ __launch_bounds__(256) void capa_attn(
    const _Float16* __restrict__ qh, const _Float16* __restrict__ kh,
    const _Float16* __restrict__ vh, const float* __restrict__ xr,
    const float* __restrict__ gamma, float* __restrict__ out)
{
    alignas(16) __shared__ _Float16 q_lds[128][QLD];
    alignas(16) __shared__ _Float16 k_lds[2][32][KLD];    // [key][ch]
    alignas(16) __shared__ _Float16 v_stage[2][32][KLD];  // [key][ch] (linear)
    alignas(16) __shared__ _Float16 v_lds[2][64][VLD];    // [ch][key] (transposed)
    __shared__ float bias_lds[NKPAD];

    const int t   = threadIdx.x;
    const int blk = blockIdx.x;
    const int bb  = blk / NWIN;
    const int wid = blk - bb * NWIN;
    const int wy  = wid / NSIDE, wx = wid - wy * NSIDE;
    const long pixbase = (long)bb * HW;

    // --- per-key additive mask, computed once per window -------------------
    for (int i = t; i < NKPAD; i += 256) {
        float bv = -1.0e30f;                         // padded keys: weight -> 0
        if (i < NKEY) {
            int dy = i / 33, dx = i - dy * 33;
            int gy = wy * SCW - SCW + dy, gx = wx * SCW - SCW + dx;
            bv = (gy >= 0 && gy < 256 && gx >= 0 && gx < 256) ? 0.0f : -99.0f;
        }
        bias_lds[i] = bv;
    }

    // --- stage Q tile (121 real queries padded to 128, zeros elsewhere) ----
    {
        int n  = t >> 1;
        int c0 = (t & 1) * 32;
        int rr = n / SCW, tt = n - rr * SCW;
        int gy = wy * SCW + rr, gx = wx * SCW + tt;
        bool valid = (n < 121) && (gy < 256) && (gx < 256);
        long src = (pixbase + gy * 256 + gx) * CH + c0;
#pragma unroll
        for (int i = 0; i < 4; ++i) {
            uint4 d = make_uint4(0u, 0u, 0u, 0u);
            if (valid) d = *(const uint4*)(qh + src + i * 8);
            *(uint4*)&q_lds[n][c0 + i * 8] = d;
        }
    }

    // --- staging helpers ---------------------------------------------------
    const int s_key = t >> 3;            // key slot 0..31
    const int s_c0  = (t & 7) * 8;       // channel group
    auto stage = [&](int ckk, int nb) {
        int kk = ckk * 32 + s_key;
        int dy = kk / 33, dx = kk - dy * 33;
        int gy = wy * SCW - SCW + dy, gx = wx * SCW - SCW + dx;
        bool valid = (kk < NKEY) && (gy >= 0) && (gy < 256) &&
                     (gx >= 0) && (gx < 256);
        if (valid) {
            long src = (pixbase + gy * 256 + gx) * CH + s_c0;
            async_copy_b128((unsigned)(size_t)&k_lds[nb][s_key][s_c0],
                            (unsigned long long)(kh + src));
            async_copy_b128((unsigned)(size_t)&v_stage[nb][s_key][s_c0],
                            (unsigned long long)(vh + src));
        } else {
            *(uint4*)&k_lds[nb][s_key][s_c0]   = make_uint4(0u, 0u, 0u, 0u);
            *(uint4*)&v_stage[nb][s_key][s_c0] = make_uint4(0u, 0u, 0u, 0u);
        }
    };
    auto xpose_v = [&](int nb) {         // v_stage [key][ch] -> v_lds [ch][key]
        uint4 d = *(const uint4*)&v_stage[nb][s_key][s_c0];
        const _Float16* e = (const _Float16*)&d;
#pragma unroll
        for (int i = 0; i < 8; ++i) v_lds[nb][s_c0 + i][s_key] = e[i];
    };

    // --- prologue: chunk 0 into buffer 0 -----------------------------------
    stage(0, 0);
    wait_asynccnt0();
    __syncthreads();
    xpose_v(0);
    __syncthreads();

    const int wave = t >> 5, lane = t & 31;
    const int low  = (lane < 16) ? 1 : 0;
    const int nq   = wave * 16 + (lane & 15);   // this lane's query column

    // Q B-fragments, loaded once (K=channel, N=query).
    v16h Bq[2];
#pragma unroll
    for (int ks = 0; ks < 2; ++ks) {
        v16h f;
        const _Float16* qrow = &q_lds[nq][0];
        int cb = ks * 32 + (low ? 0 : 16);
#pragma unroll
        for (int j = 0; j < 8; ++j) {
            f[2 * j]     = qrow[cb + 2 * j];
            f[2 * j + 1] = qrow[cb + 2 * j + 1];
        }
        Bq[ks] = f;
    }

    v8f racc[4] = {};                 // 64 output channels x 16 queries
    float m_run = -3.0e38f, l_run = 0.0f;

    for (int ck = 0; ck < NCHUNK; ++ck) {
        const int nb = ck & 1;
        const int kb = ck * 32;

        // prefetch next chunk into the other buffer (overlaps with compute)
        if (ck + 1 < NCHUNK) stage(ck + 1, nb ^ 1);

        // ---- scores: two 16(key) x 16(query) tiles, contract c=64 ----
        v8f S[2];
#pragma unroll
        for (int kt = 0; kt < 2; ++kt) {
            v8f s = {};
#pragma unroll
            for (int ks = 0; ks < 2; ++ks) {
                v16h a;
                const _Float16* kr = &k_lds[nb][kt * 16 + (lane & 15)][0];
#pragma unroll
                for (int p = 0; p < 8; ++p) {
                    int K0 = ks * 32 + 2 * p
                             + (low ? 0 : 16);
                    a[2 * p]     = kr[K0];
                    a[2 * p + 1] = kr[K0 + 1];
                }
                s = __builtin_amdgcn_wmma_f32_16x16x32_f16(
                        false, a, false, Bq[ks], (short)0, s, false, false);
            }
            S[kt] = s;
        }

        // ---- additive mask from precomputed per-key bias ----
#pragma unroll
        for (int kt = 0; kt < 2; ++kt) {
            const float* bl = &bias_lds[kb + kt * 16 + (low ? 0 : 8)];
#pragma unroll
            for (int r = 0; r < 8; ++r) S[kt][r] += bl[r];
        }

        // ---- online softmax (per-query stats replicated in lane pairs) ----
        float cm = S[0][0];
#pragma unroll
        for (int r = 1; r < 8; ++r) cm = fmaxf(cm, S[0][r]);
#pragma unroll
        for (int r = 0; r < 8; ++r) cm = fmaxf(cm, S[1][r]);
        cm = fmaxf(cm, __shfl_xor(cm, 16));
        float m_new = fmaxf(m_run, cm);
        float corr  = __expf(m_run - m_new);
        float rs = 0.0f;
#pragma unroll
        for (int kt = 0; kt < 2; ++kt)
#pragma unroll
            for (int r = 0; r < 8; ++r) {
                float e = __expf(S[kt][r] - m_new);
                S[kt][r] = e;
                rs += e;
            }
        rs += __shfl_xor(rs, 16);
        l_run = l_run * corr + rs;
        m_run = m_new;
#pragma unroll
        for (int mt = 0; mt < 4; ++mt)
#pragma unroll
            for (int r = 0; r < 8; ++r) racc[mt][r] *= corr;

        // ---- build P as a B fragment (K=key, N=query), f16 ----
        v16h bp;
#pragma unroll
        for (int j = 0; j < 4; ++j) {
            float oa = __shfl_xor(S[1][2 * j], 16);
            float ob = __shfl_xor(S[1][2 * j + 1], 16);
            bp[2 * j]     = (_Float16)(low ? S[0][2 * j]     : oa);
            bp[2 * j + 1] = (_Float16)(low ? S[0][2 * j + 1] : ob);
        }
#pragma unroll
        for (int j = 4; j < 8; ++j) {
            float oa = __shfl_xor(S[0][2 * j - 8], 16);
            float ob = __shfl_xor(S[0][2 * j - 7], 16);
            bp[2 * j]     = (_Float16)(low ? oa : S[1][2 * j - 8]);
            bp[2 * j + 1] = (_Float16)(low ? ob : S[1][2 * j - 7]);
        }

        // ---- r += V(64ch x 32key) @ P(32key x 16q) ----
#pragma unroll
        for (int mt = 0; mt < 4; ++mt) {
            v16h a;
            const _Float16* vr = &v_lds[nb][mt * 16 + (lane & 15)][0];
#pragma unroll
            for (int p = 0; p < 8; ++p) {
                int K0 = 2 * p + (low ? 0 : 16);
                a[2 * p]     = vr[K0];
                a[2 * p + 1] = vr[K0 + 1];
            }
            racc[mt] = __builtin_amdgcn_wmma_f32_16x16x32_f16(
                    false, a, false, bp, (short)0, racc[mt], false, false);
        }

        // ---- handoff: finish async copies, transpose prefetched V ----
        wait_asynccnt0();
        __syncthreads();
        if (ck + 1 < NCHUNK) xpose_v(nb ^ 1);
        __syncthreads();
    }

    // ---- epilogue: out = x_ref + gamma * (r / l) on valid pixels ----
    float inv = (l_run > 0.0f) ? 1.0f / l_run : 0.0f;
    float g = gamma[0];
    if (nq < 121) {
        int rr = nq / SCW, tt = nq - rr * SCW;
        int gy = wy * SCW + rr, gx = wx * SCW + tt;
        if (gy < 256 && gx < 256) {
            long sp = (long)gy * 256 + gx;
#pragma unroll
            for (int mt = 0; mt < 4; ++mt) {
                int ch0 = mt * 16 + (low ? 0 : 8);
#pragma unroll
                for (int r = 0; r < 8; ++r) {
                    long o = ((long)bb * CH + ch0 + r) * HW + sp;
                    out[o] = xr[o] + g * (racc[mt][r] * inv);
                }
            }
        }
    }
}

// ---------------------------------------------------------------------------
extern "C" void kernel_launch(void* const* d_in, const int* in_sizes, int n_in,
                              void* d_out, int out_size, void* d_ws,
                              size_t ws_size, hipStream_t stream)
{
    (void)in_sizes; (void)n_in; (void)out_size; (void)ws_size;
    const float* xk    = (const float*)d_in[0];
    const float* xr    = (const float*)d_in[1];
    const float* Wq    = (const float*)d_in[2];
    const float* Wk    = (const float*)d_in[3];
    const float* Wv    = (const float*)d_in[4];
    const float* gamma = (const float*)d_in[5];

    const size_t per = (size_t)BATCH * HW * CH;    // f16 elems per map
    _Float16* qh = (_Float16*)d_ws;
    _Float16* kh = qh + per;
    _Float16* vh = kh + per;

    // 131072 pixels / 128 per block
    capa_proj<<<1024, 256, 0, stream>>>(xk, xr, Wq, Wk, Wv, qh, kh, vh);
    // one block per (batch, window)
    capa_attn<<<BATCH * NWIN, 256, 0, stream>>>(qh, kh, vh, xr, gamma,
                                                (float*)d_out);
}
